// QuaternionLinear_7189775253858
// MI455X (gfx1250) — compile-verified, matrix operand order checked by reading, not run on hardware
//
#include <hip/hip_runtime.h>

// ---------------------------------------------------------------------------
// QuaternionLinear as one dense GEMM: Y(32768x1024) = X(32768x1024) * M(1024x1024) + bias
// M[(i*4+ci),(o*4+co)] = sign[ci][co] * weight[o, i, comp[ci][co]]
// bf16 hi/lo split (3x V_WMMA_F32_16X16X32_BF16) for ~f32 accuracy at XDL rate.
// A staged to LDS with GLOBAL_LOAD_ASYNC_TO_LDS_B128 + double buffering.
// B pre-laid out in d_ws in per-lane fragment order; inner loop uses one base
// pointer + immediate offsets, with software double-buffered B fragments.
// ---------------------------------------------------------------------------

typedef __attribute__((ext_vector_type(16))) __bf16 v16bf;
typedef __attribute__((ext_vector_type(8)))  float  v8f;
typedef int v4i __attribute__((vector_size(16)));

#define MDIM   32768
#define KDIM   1024
#define NDIM   1024
#define MTILE  128      // rows per workgroup
#define NTILE  128      // cols per workgroup (8 subtiles of 16)
#define KCHUNK 32       // bf16 WMMA K
#define NITER  (KDIM / KCHUNK)
#define LDSPITCH 36     // padded row pitch (floats) for the A stage buffer
#define BLO_OFF (2u * 1024u * 1024u)   // byte offset of the lo-plane inside d_ws
#define BSTEP_K 65536                  // bytes of B fragments per kc step (64*1024)

#if __has_builtin(__builtin_amdgcn_global_load_async_to_lds_b128) && \
    __has_builtin(__builtin_amdgcn_s_wait_asynccnt)
#define USE_ASYNC_LDS 1
#else
#define USE_ASYNC_LDS 0
#endif

__device__ __forceinline__ unsigned short f2bf(float f) {
    __bf16 h = (__bf16)f;              // RNE
    unsigned short u;
    __builtin_memcpy(&u, &h, 2);
    return u;
}
__device__ __forceinline__ float bf2f(unsigned short u) {
    __bf16 h;
    __builtin_memcpy(&h, &u, 2);
    return (float)h;
}

// ---------------------------------------------------------------------------
// Phase 1: expand weight (256,256,4) into dense M, pre-split into bf16 hi/lo,
// stored directly in per-lane WMMA B-fragment order:
//   fragment block (kc, nt): 32 lanes x 8 dwords (16 bf16 K-values per lane)
//   lane L -> column n = nt*16 + (L%16); dword j -> K pair starting at
//   k = kc*32 + (j>=4 ? 16 : 0) + (L/16)*8 + (j%4)*2    (mirrors 16-bit A layout)
//   byte offset = kc*65536 + nt*1024 + L*32 + j*4
// ---------------------------------------------------------------------------
__global__ __launch_bounds__(256) void quat_expand_weight(
        const float* __restrict__ w,
        unsigned int* __restrict__ bhi,
        unsigned int* __restrict__ blo) {
    const int comp[4][4] = {{0,1,2,3},{1,0,3,2},{2,3,0,1},{3,2,1,0}};
    const float sgn[4][4] = {{ 1.f, 1.f, 1.f, 1.f},
                             {-1.f, 1.f, 1.f,-1.f},
                             {-1.f,-1.f, 1.f, 1.f},
                             {-1.f, 1.f,-1.f, 1.f}};
    int tid  = blockIdx.x * 256 + threadIdx.x;   // 65536 threads total
    int lane = tid & 31;
    int nt   = (tid >> 5) & 63;                  // N tile (16 cols each)
    int kc   = tid >> 11;                        // K chunk (32 each)
    int half = lane >> 4;
    int n    = nt * 16 + (lane & 15);
    int o    = n >> 2;
    int co   = n & 3;

    unsigned int hi[8], lo[8];
#pragma unroll
    for (int j = 0; j < 8; ++j) {
        int kb = kc * 32 + ((j >= 4) ? 16 : 0) + half * 8 + (j & 3) * 2;
        float v0, v1;
        {
            int k = kb;     int ci = k & 3; int i = k >> 2;
            v0 = sgn[ci][co] * w[o * 1024 + i * 4 + comp[ci][co]];
        }
        {
            int k = kb + 1; int ci = k & 3; int i = k >> 2;
            v1 = sgn[ci][co] * w[o * 1024 + i * 4 + comp[ci][co]];
        }
        unsigned short h0 = f2bf(v0), h1 = f2bf(v1);
        hi[j] = (unsigned)h0 | ((unsigned)h1 << 16);
        unsigned short l0 = f2bf(v0 - bf2f(h0));
        unsigned short l1 = f2bf(v1 - bf2f(h1));
        lo[j] = (unsigned)l0 | ((unsigned)l1 << 16);
    }
    size_t base = ((size_t)(kc * 64 + nt) * 32 + (size_t)lane) * 8;
    uint4* ph = (uint4*)(bhi + base);
    ph[0] = make_uint4(hi[0], hi[1], hi[2], hi[3]);
    ph[1] = make_uint4(hi[4], hi[5], hi[6], hi[7]);
    uint4* pl = (uint4*)(blo + base);
    pl[0] = make_uint4(lo[0], lo[1], lo[2], lo[3]);
    pl[1] = make_uint4(lo[4], lo[5], lo[6], lo[7]);
}

// ---------------------------------------------------------------------------
// A-chunk staging: each thread moves 64 contiguous bytes (4 x b128).
// Async path: memory -> LDS directly via the ASYNCcnt-tracked DMA path.
// Builtin signature (probe-confirmed): (v4i AS1*, v4i AS3*, imm offset, imm cpol)
// ---------------------------------------------------------------------------
__device__ __forceinline__ void stage_chunk(const float* g, float* l) {
#if USE_ASYNC_LDS
    typedef __attribute__((address_space(1))) v4i GV4;
    typedef __attribute__((address_space(3))) v4i LV4;
    GV4* gp = (GV4*)(const_cast<float*>(g));
    LV4* lp = (LV4*)l;
    __builtin_amdgcn_global_load_async_to_lds_b128(gp, lp, 0, 0);
    __builtin_amdgcn_global_load_async_to_lds_b128(gp, lp, 16, 0);
    __builtin_amdgcn_global_load_async_to_lds_b128(gp, lp, 32, 0);
    __builtin_amdgcn_global_load_async_to_lds_b128(gp, lp, 48, 0);
#else
    const float4* gv = (const float4*)g;
    float4 t0 = gv[0], t1 = gv[1], t2 = gv[2], t3 = gv[3];
    float4* lv = (float4*)l;
    lv[0] = t0; lv[1] = t1; lv[2] = t2; lv[3] = t3;
#endif
}

// ---------------------------------------------------------------------------
// Main GEMM kernel: 8 waves, 128x128 tile, K streamed in chunks of 32,
// double-buffered LDS A staging, double-buffered B fragments.
// ---------------------------------------------------------------------------
union AFrag { unsigned int w[8]; v16bf v; };
union BFrag { uint4 q[2]; v16bf v; };

__device__ __forceinline__ void loadB(const char* pB, int s, BFrag& Bh, BFrag& Bl) {
    // s is a compile-time constant in the unrolled loop -> immediate offsets
    Bh.q[0] = *(const uint4*)(pB + s * 1024);
    Bh.q[1] = *(const uint4*)(pB + s * 1024 + 16);
    Bl.q[0] = *(const uint4*)(pB + BLO_OFF + s * 1024);
    Bl.q[1] = *(const uint4*)(pB + BLO_OFF + s * 1024 + 16);
}

__global__ __launch_bounds__(256) void quat_gemm(
        const float* __restrict__ x,
        const unsigned int* __restrict__ bhi,
        const float* __restrict__ bias,
        float* __restrict__ out) {
    __shared__ float Alds[2][MTILE * LDSPITCH];   // 2 x 18 KB

    const int tid  = threadIdx.x;
    const int wave = tid >> 5;        // 0..7 -> owns rows [wave*16, wave*16+16)
    const int lane = tid & 31;
    const int half = lane >> 4;
    const int nl   = lane & 15;
    const int blockN = blockIdx.x;    // 0..7
    const int blockM = blockIdx.y;    // 0..255

    v8f acc[8];
#pragma unroll
    for (int s = 0; s < 8; ++s)
#pragma unroll
        for (int j = 0; j < 8; ++j) acc[s][j] = 0.0f;

    // staging assignment: thread -> (row = tid/2, 64B half-row chosen by tid&1)
    const int   srow = tid >> 1;            // 0..127
    const int   scol = (tid & 1) * 16;      // 0 or 16
    const float* gsrc = x + (size_t)(blockM * MTILE + srow) * KDIM + scol;
    float* ldst[2] = { &Alds[0][srow * LDSPITCH + scol],
                       &Alds[1][srow * LDSPITCH + scol] };

    const float* arow[2] = { &Alds[0][(wave * 16 + nl) * LDSPITCH],
                             &Alds[1][(wave * 16 + nl) * LDSPITCH] };

    // per-lane B base pointer; everything else is immediate offsets
    const char* pB = (const char*)bhi + (size_t)blockN * 8192 + (size_t)lane * 32;

    stage_chunk(gsrc, ldst[0]);            // prologue: stage chunk 0 into buffer 0
    const float* gstage = gsrc + KCHUNK;   // next chunk's global source

    for (int kc = 0; kc < NITER; ++kc) {
        const int cur = kc & 1;
        if (kc + 1 < NITER) {
            // issue next chunk into the other buffer (its last readers were
            // fenced by the end-of-compute barrier of iteration kc-1)
            stage_chunk(gstage, ldst[cur ^ 1]);
            gstage += KCHUNK;
        }

        // prefetch this chunk's first B fragments while the async stage and
        // barrier resolve (B is plain global memory, independent of LDS)
        BFrag Bh[2], Bl[2];
        loadB(pB, 0, Bh[0], Bl[0]);

#if USE_ASYNC_LDS
        if (kc + 1 < NITER) __builtin_amdgcn_s_wait_asynccnt(4);  // cur chunk landed
        else                __builtin_amdgcn_s_wait_asynccnt(0);
#endif
        __syncthreads();   // current buffer visible to all waves

        // per-lane A fragment (documented 16-bit A layout, wave32):
        // dwords 0..3 = K in [half*8, half*8+8), dwords 4..7 = K in [16+half*8, ...)
        const float* ar = arow[cur];
        union { float4 v4[4]; float f[16]; } A;
        A.v4[0] = *(const float4*)(ar + half * 8 + 0);
        A.v4[1] = *(const float4*)(ar + half * 8 + 4);
        A.v4[2] = *(const float4*)(ar + 16 + half * 8 + 0);
        A.v4[3] = *(const float4*)(ar + 16 + half * 8 + 4);

        AFrag Ah, Al;
#pragma unroll
        for (int j = 0; j < 8; ++j) {
            float f0 = A.f[2 * j], f1 = A.f[2 * j + 1];
            unsigned short h0 = f2bf(f0), h1 = f2bf(f1);
            Ah.w[j] = (unsigned)h0 | ((unsigned)h1 << 16);
            unsigned short l0 = f2bf(f0 - bf2f(h0));
            unsigned short l1 = f2bf(f1 - bf2f(h1));
            Al.w[j] = (unsigned)l0 | ((unsigned)l1 << 16);
        }

#pragma unroll
        for (int s = 0; s < 8; ++s) {
            const int c = s & 1;
            if (s + 1 < 8) loadB(pB, s + 1, Bh[c ^ 1], Bl[c ^ 1]);  // hide L2 latency
            // D += Ahi*Bhi + Alo*Bhi + Ahi*Blo  (bf16x2 split, f32 accumulate)
            acc[s] = __builtin_amdgcn_wmma_f32_16x16x32_bf16(
                false, Ah.v, false, Bh[c].v, (short)0, acc[s], false, false);
            acc[s] = __builtin_amdgcn_wmma_f32_16x16x32_bf16(
                false, Al.v, false, Bh[c].v, (short)0, acc[s], false, false);
            acc[s] = __builtin_amdgcn_wmma_f32_16x16x32_bf16(
                false, Ah.v, false, Bl[c].v, (short)0, acc[s], false, false);
        }
        pB += BSTEP_K;     // advance to next K-chunk's fragment blocks
        __syncthreads();   // all reads of current buffer done before re-staging
    }

    // Epilogue: f32 C/D layout -> lane holds (M = j + 8*half, N = lane%16).
    // C has no reuse: non-temporal stores keep L2 for A/B.
    const int mbase = blockM * MTILE + wave * 16 + half * 8;
#pragma unroll
    for (int s = 0; s < 8; ++s) {
        const int n = (blockN * 8 + s) * 16 + nl;
        const float bv = bias[n];
#pragma unroll
        for (int j = 0; j < 8; ++j) {
            __builtin_nontemporal_store(acc[s][j] + bv,
                                        &out[(size_t)(mbase + j) * NDIM + n]);
        }
    }
}

extern "C" void kernel_launch(void* const* d_in, const int* in_sizes, int n_in,
                              void* d_out, int out_size, void* d_ws, size_t ws_size,
                              hipStream_t stream) {
    (void)in_sizes; (void)n_in; (void)out_size; (void)ws_size;
    const float* x    = (const float*)d_in[0];   // (32768, 256, 4) f32
    const float* w    = (const float*)d_in[1];   // (256, 256, 4)  f32
    const float* bias = (const float*)d_in[2];   // (256, 4)       f32
    float* out = (float*)d_out;                  // (32768, 256, 4) f32

    unsigned int* bhi = (unsigned int*)d_ws;             // 2 MiB (1024*1024 bf16)
    unsigned int* blo = bhi + (KDIM * NDIM / 2);         // 2 MiB, at bhi + BLO_OFF

    quat_expand_weight<<<256, 256, 0, stream>>>(w, bhi, blo);

    dim3 grid(NDIM / NTILE, MDIM / MTILE);   // (8, 256), N fastest -> A reuse in L2
    quat_gemm<<<grid, 256, 0, stream>>>(x, bhi, bias, out);
}
